// GCNEncoder_23922967838756
// MI455X (gfx1250) — matrix operand order repaired and checked by reading, not compiled, not run on hardware
//
#include <hip/hip_runtime.h>

typedef __attribute__((ext_vector_type(16))) __bf16 v16bf;
typedef __attribute__((ext_vector_type(4)))  __bf16 v4bf;
typedef __attribute__((ext_vector_type(8)))  float  v8f;

// ---------------- utility: vectorized zero fill ----------------
__global__ __launch_bounds__(256) void zero_f4(float4* __restrict__ p, int n4) {
    int i = blockIdx.x * 256 + threadIdx.x;
    if (i < n4) p[i] = make_float4(0.f, 0.f, 0.f, 0.f);
}

// ---------------- degree accumulation ----------------
__global__ __launch_bounds__(256) void degree_kernel(const int* __restrict__ row,
                                                     float* __restrict__ deg, int E) {
    int e = blockIdx.x * 256 + threadIdx.x;
    if (e < E)
        __hip_atomic_fetch_add(deg + row[e], 1.0f, __ATOMIC_RELAXED, __HIP_MEMORY_SCOPE_AGENT);
}

// 4-wide loaders (f32 and bf16 sources), always producing f32 values
__device__ inline void load4(const float* __restrict__ p, float& a, float& b, float& c, float& d) {
    float4 v = *(const float4*)p;
    a = v.x; b = v.y; c = v.z; d = v.w;
}
__device__ inline void load4(const __bf16* __restrict__ p, float& a, float& b, float& c, float& d) {
    v4bf v = *(const v4bf*)p;
    a = (float)v[0]; b = (float)v[1]; c = (float)v[2]; d = (float)v[3];
}

// ---------------- edge scatter-add: dst[row] += src[col], D elems/row ----------------
template<int D, typename ST>
__global__ __launch_bounds__(256) void scatter_add(const int* __restrict__ row,
                                                   const int* __restrict__ col,
                                                   const ST* __restrict__ src,
                                                   float* __restrict__ dst, int nwork) {
    constexpr int C4 = D / 4;               // 4-element chunks per row
    int idx = blockIdx.x * 256 + threadIdx.x;
    if (idx >= nwork) return;
    int e = idx / C4;
    int c = (idx % C4) * 4;
    int r = row[e];
    int s = col[e];
    float v0, v1, v2, v3;
    load4(src + (size_t)s * D + c, v0, v1, v2, v3);
    float* p = dst + (size_t)r * D + c;
    __hip_atomic_fetch_add(p + 0, v0, __ATOMIC_RELAXED, __HIP_MEMORY_SCOPE_AGENT);
    __hip_atomic_fetch_add(p + 1, v1, __ATOMIC_RELAXED, __HIP_MEMORY_SCOPE_AGENT);
    __hip_atomic_fetch_add(p + 2, v2, __ATOMIC_RELAXED, __HIP_MEMORY_SCOPE_AGENT);
    __hip_atomic_fetch_add(p + 3, v3, __ATOMIC_RELAXED, __HIP_MEMORY_SCOPE_AGENT);
}

// ---------------- pre-swizzle W (f32, [IN_DIM x OUT_DIM] row-major) into WMMA
// B-operand bf16 layout. Tile (kc,t) covers K rows kc*32..+31, cols t*16..+15.
// CDNA5 ISA B 32x16 bf16 (wave32): lane l holds n = l&15,
// element j (of 16 bf16) holds k = (j&7) + (j>=8 ? 16 : 0) + (l>=16 ? 8 : 0). ----------------
template<int IN_DIM, int OUT_DIM>
__global__ __launch_bounds__(256) void swizzle_w(const float* __restrict__ W,
                                                 __bf16* __restrict__ Wswz) {
    constexpr int KC = IN_DIM / 32;
    constexpr int NT = OUT_DIM / 16;
    int idx = blockIdx.x * 256 + threadIdx.x;
    if (idx >= KC * NT * 32 * 16) return;
    int j    = idx & 15;
    int lane = (idx >> 4) & 31;
    int tile = idx >> 9;              // kc*NT + t
    int t    = tile % NT;
    int kc   = tile / NT;
    int n = t * 16 + (lane & 15);
    int k = kc * 32 + (j & 7) + ((j >= 8) ? 16 : 0) + ((lane >= 16) ? 8 : 0);
    Wswz[idx] = (__bf16)W[(size_t)k * OUT_DIM + n];
}

// ---------------- fused GCN dense layer: out = relu?((agg/deg + xin) @ W + b)
// One wave computes a 16-row x OUT_DIM output strip with bf16 WMMA, f32 accum.
// A operand 16x32 bf16 layout (ISA): lane l holds m = l&15, element j holds
// k = kbase + j + (j>=8 ? 8 : 0) with kbase = kc*32 + (l>=16 ? 8 : 0).
// C/D layout: VGPR j -> m = j + (l>=16 ? 8 : 0), n = l&15. ----------------
template<int IN_DIM, int OUT_DIM, bool RELU, typename XT, typename OT>
__global__ __launch_bounds__(128) void gcn_gemm(const float* __restrict__ agg,
                                                const XT* __restrict__ xin,
                                                const float* __restrict__ deg,
                                                const __bf16* __restrict__ Wswz,
                                                const float* __restrict__ bias,
                                                OT* __restrict__ out, int nrows) {
    constexpr int KC = IN_DIM / 32;
    constexpr int NT = OUT_DIM / 16;
    const int wave = threadIdx.x >> 5;
    const int lane = threadIdx.x & 31;
    const int m0 = (blockIdx.x * 4 + wave) * 16;
    if (m0 >= nrows) return;

    const int   mrow = m0 + (lane & 15);
    const float dinv = 1.0f / fmaxf(deg[mrow], 1.0f);
    const float* aggr = agg + (size_t)mrow * IN_DIM;
    const XT*    xr   = xin + (size_t)mrow * IN_DIM;

    v8f acc[NT];
#pragma unroll
    for (int t = 0; t < NT; ++t) acc[t] = (v8f){0.f,0.f,0.f,0.f,0.f,0.f,0.f,0.f};

#pragma unroll
    for (int kc = 0; kc < KC; ++kc) {
        const int kbase = kc * 32 + ((lane >> 4) << 3);
        v16bf a;
#pragma unroll
        for (int j = 0; j < 8; ++j)
            a[j]     = (__bf16)(aggr[kbase +      j] * dinv + (float)xr[kbase +      j]);
#pragma unroll
        for (int j = 0; j < 8; ++j)
            a[8 + j] = (__bf16)(aggr[kbase + 16 + j] * dinv + (float)xr[kbase + 16 + j]);
#pragma unroll
        for (int t = 0; t < NT; ++t) {
            const v16bf b = *(const v16bf*)(Wswz + ((size_t)(kc * NT + t) * 32 + lane) * 16);
            acc[t] = __builtin_amdgcn_wmma_f32_16x16x32_bf16(
                /*neg_a=*/false, a, /*neg_b=*/false, b,
                /*c_mod=*/(short)0, acc[t], /*reuse_a=*/false, /*reuse_b=*/false);
        }
    }

    const int nlo = lane & 15;
    const int mhi = (lane >> 4) * 8;
#pragma unroll
    for (int t = 0; t < NT; ++t) {
        const int n = t * 16 + nlo;
        const float bv = bias[n];
#pragma unroll
        for (int j = 0; j < 8; ++j) {
            float v = acc[t][j] + bv;
            if (RELU) v = fmaxf(v, 0.f);
            out[(size_t)(m0 + mhi + j) * OUT_DIM + n] = (OT)v;
        }
    }
}

extern "C" void kernel_launch(void* const* d_in, const int* in_sizes, int n_in,
                              void* d_out, int out_size, void* d_ws, size_t ws_size,
                              hipStream_t stream) {
    const float* x  = (const float*)d_in[0];
    const int*   ei = (const int*)  d_in[1];   // [2, E] int32
    const float* W1 = (const float*)d_in[2];
    const float* b1 = (const float*)d_in[3];
    const float* W2 = (const float*)d_in[4];
    const float* b2 = (const float*)d_in[5];
    float* out = (float*)d_out;

    const int N = in_sizes[0] / 64;
    const int E = in_sizes[1] / 2;
    const int* row = ei;
    const int* col = ei + E;

    // workspace layout: [deg N f32][agg N*128 f32 (layer1 uses first N*64)]
    //                   [h1 N*128 bf16][W1s/W2s bf16]
    float*  deg = (float*)d_ws;
    float*  agg = deg + N;
    __bf16* h1  = (__bf16*)(agg + (size_t)N * 128);
    __bf16* W1s = h1 + (size_t)N * 128;
    __bf16* W2s = W1s + 64 * 128;

    // weight swizzles (independent, launch first)
    swizzle_w<64, 128><<<(64 * 128 + 255) / 256, 256, 0, stream>>>(W1, W1s);
    swizzle_w<128, 64><<<(128 * 64 + 255) / 256, 256, 0, stream>>>(W2, W2s);

    // ---- layer 1 ----
    {   // zero deg + agg1 (contiguous: N + N*64 floats)
        int n4 = (N + N * 64) / 4;
        zero_f4<<<(n4 + 255) / 256, 256, 0, stream>>>((float4*)deg, n4);
    }
    degree_kernel<<<(E + 255) / 256, 256, 0, stream>>>(row, deg, E);
    {
        int nwork = E * (64 / 4);
        scatter_add<64, float><<<(nwork + 255) / 256, 256, 0, stream>>>(row, col, x, agg, nwork);
    }
    {
        int nblk = ((N / 16) + 3) / 4;
        gcn_gemm<64, 128, true, float, __bf16><<<nblk, 128, 0, stream>>>(agg, x, deg, W1s, b1, h1, N);
    }

    // ---- layer 2 ----
    {   // zero agg2 (N*128 floats); safe to reuse: gemm1 already consumed agg1 in stream order
        int n4 = (N * 128) / 4;
        zero_f4<<<(n4 + 255) / 256, 256, 0, stream>>>((float4*)agg, n4);
    }
    {
        int nwork = E * (128 / 4);
        scatter_add<128, __bf16><<<(nwork + 255) / 256, 256, 0, stream>>>(row, col, h1, agg, nwork);
    }
    {
        int nblk = ((N / 16) + 3) / 4;
        gcn_gemm<128, 64, false, __bf16, float><<<nblk, 128, 0, stream>>>(agg, h1, deg, W2s, b2, out, N);
    }
}
